// MixtralSparseMoeBlock_69243462746561
// MI455X (gfx1250) — compile-verified, matrix operand order checked by reading, not run on hardware
//
#include <hip/hip_runtime.h>
#include <stdint.h>

// Problem constants (MixtralSparseMoeBlock): T=8192, H=1024, I=3584, E=8, K=2
#define T_TOK 8192
#define HID   1024
#define INT_I 3584
#define NEXP  8

// GEMM tiling: 64x128 C-tile per block, 256 threads = 8 wave32 waves arranged
// 4(M) x 2(N); each wave owns a 16x64 strip = four 16x16 WMMA tiles and reuses
// one A fragment across them. BK=64 = two bf16-WMMA K-steps per LDS stage.
// LDS stages are double-buffered; async DMA for stage i+1 overlaps WMMA on i.
#define BM 64
#define BN 128
#define BK 64

typedef unsigned short ushort_t;
typedef __attribute__((ext_vector_type(16))) __bf16 v16bf;
typedef __attribute__((ext_vector_type(8)))  float  v8f;
typedef int vint4 __attribute__((vector_size(4 * sizeof(int))));

union Frag {
  uint4 u[2];   // 2 x 16B = 32B
  v16bf v;      // 16 bf16 = 32B (one WMMA A/B fragment per lane)
};

// fp32 -> bf16 round-to-nearest-even
__device__ __forceinline__ ushort_t f2bf(float f) {
  unsigned u = __float_as_uint(f);
  u += 0x7FFFu + ((u >> 16) & 1u);
  return (ushort_t)(u >> 16);
}
__device__ __forceinline__ unsigned pack2(float lo, float hi) {
  return (unsigned)f2bf(lo) | ((unsigned)f2bf(hi) << 16);
}

// ---- CDNA5 async global->LDS staging (ASYNCcnt path), probe-verified on this
// ---- toolchain; guarded so compile cannot break elsewhere.
#if defined(__gfx1250__) && __has_builtin(__builtin_amdgcn_global_load_async_to_lds_b128) && __has_builtin(__builtin_amdgcn_s_wait_asynccnt)
#define USE_ASYNC_LDS 1
#else
#define USE_ASYNC_LDS 0
#endif

__device__ __forceinline__ void cp16_g2l(const void* g, void* l) {
#if USE_ASYNC_LDS
  __builtin_amdgcn_global_load_async_to_lds_b128(
      (__attribute__((address_space(1))) vint4*)g,
      (__attribute__((address_space(3))) vint4*)l,
      0, 0);
#else
  *(uint4*)l = *(const uint4*)g;
#endif
}
// Publish: my async ops done, then block-wide barrier. After this returns,
// (a) the just-awaited stage is visible to all waves, and (b) every wave has
// finished its previous compute phase -> the *other* buffer is reusable.
__device__ __forceinline__ void publish_stage() {
#if USE_ASYNC_LDS
  __builtin_amdgcn_s_wait_asynccnt(0);
#endif
  __syncthreads();
}

// -------------------------------------------------------------------------
// Utility kernels
// -------------------------------------------------------------------------
__global__ __launch_bounds__(256) void zero_f32(float* __restrict__ p, int n) {
  int i = blockIdx.x * 256 + threadIdx.x;
  if (i < n) p[i] = 0.f;
}

// Vectorized fp32 -> bf16 cast: 8 elements per thread (2x b128 load, 1x b128 store)
__global__ __launch_bounds__(256) void cast_bf16x8(const float* __restrict__ s,
                                                   ushort_t* __restrict__ d, int n8) {
  int i = blockIdx.x * 256 + threadIdx.x;
  if (i >= n8) return;
  const float4* sp = (const float4*)(s + (size_t)i * 8);
  float4 a0 = sp[0], a1 = sp[1];
  uint4 p;
  p.x = pack2(a0.x, a0.y); p.y = pack2(a0.z, a0.w);
  p.z = pack2(a1.x, a1.y); p.w = pack2(a1.z, a1.w);
  *(uint4*)(d + (size_t)i * 8) = p;
}

// -------------------------------------------------------------------------
// Router: logits = x @ gate_w^T + gate_b ; softmax fp32 ; top-2 (NO renorm);
// writes dense combine weights [T, E].
// -------------------------------------------------------------------------
__global__ __launch_bounds__(256) void moe_router(const float* __restrict__ x,
                                                  const float* __restrict__ gw,
                                                  const float* __restrict__ gb,
                                                  float* __restrict__ combine) {
  __shared__ float sg[NEXP * HID];  // 32 KB: whole gate matrix in LDS
  for (int i = threadIdx.x; i < NEXP * HID; i += 256) sg[i] = gw[i];
  __syncthreads();

  const int t = blockIdx.x * 256 + threadIdx.x;  // one token per thread
  float acc[NEXP];
#pragma unroll
  for (int e = 0; e < NEXP; ++e) acc[e] = gb[e];
  const float* xr = x + (size_t)t * HID;
  for (int h = 0; h < HID; ++h) {
    const float xv = xr[h];
#pragma unroll
    for (int e = 0; e < NEXP; ++e) acc[e] += xv * sg[e * HID + h];
  }
  // softmax (fp32)
  float mx = acc[0];
#pragma unroll
  for (int e = 1; e < NEXP; ++e) mx = fmaxf(mx, acc[e]);
  float p[NEXP];
  float sum = 0.f;
#pragma unroll
  for (int e = 0; e < NEXP; ++e) { p[e] = __expf(acc[e] - mx); sum += p[e]; }
  const float inv = 1.f / sum;
  // top-2, lower index wins ties (matches lax.top_k)
  int i0 = 0; float v0 = p[0];
#pragma unroll
  for (int e = 1; e < NEXP; ++e) if (p[e] > v0) { v0 = p[e]; i0 = e; }
  int i1 = -1; float v1 = -1.f;
#pragma unroll
  for (int e = 0; e < NEXP; ++e) if (e != i0 && p[e] > v1) { v1 = p[e]; i1 = e; }

  float* cr = combine + (size_t)t * NEXP;
#pragma unroll
  for (int e = 0; e < NEXP; ++e)
    cr[e] = (e == i0) ? v0 * inv : ((e == i1) ? v1 * inv : 0.f);
}

// -------------------------------------------------------------------------
// GEMM1 helpers
// -------------------------------------------------------------------------
__device__ __forceinline__ void stage1(const ushort_t* __restrict__ xb,
                                       const ushort_t* __restrict__ w1b,
                                       const ushort_t* __restrict__ w3b,
                                       int rowBase, int colBase, int kk, int tid,
                                       ushort_t (*sA)[BK], ushort_t (*sB1)[BK],
                                       ushort_t (*sB3)[BK]) {
#pragma unroll
  for (int i = 0; i < 2; ++i) {
    const int c = tid + i * 256;
    const int r = c >> 3, col = (c & 7) * 8;
    cp16_g2l(xb + (size_t)(rowBase + r) * HID + kk + col, &sA[r][col]);
  }
#pragma unroll
  for (int i = 0; i < 4; ++i) {
    const int c = tid + i * 256;
    const int r = c >> 3, col = (c & 7) * 8;
    cp16_g2l(w1b + (size_t)(colBase + r) * HID + kk + col, &sB1[r][col]);
    cp16_g2l(w3b + (size_t)(colBase + r) * HID + kk + col, &sB3[r][col]);
  }
}

__device__ __forceinline__ void compute1(const ushort_t (*sA)[BK],
                                         const ushort_t (*sB1)[BK],
                                         const ushort_t (*sB3)[BK],
                                         int m0, int nb, int hlf, int l16,
                                         v8f* acc1, v8f* acc3) {
#pragma unroll
  for (int ks = 0; ks < 2; ++ks) {       // two WMMA K-steps per stage
    const int kc = ks * 32;
    Frag fa;                              // A fragment reused across 8 WMMAs
    fa.u[0] = *(const uint4*)&sA[m0 + l16][kc + hlf * 8];
    fa.u[1] = *(const uint4*)&sA[m0 + l16][kc + 16 + hlf * 8];
    Frag b1[4], b3[4];                    // preload -> one ds clause, one wait
#pragma unroll
    for (int nt = 0; nt < 4; ++nt) {
      const int n = nb + nt * 16 + l16;
      b1[nt].u[0] = *(const uint4*)&sB1[n][kc + hlf * 8];
      b1[nt].u[1] = *(const uint4*)&sB1[n][kc + 16 + hlf * 8];
      b3[nt].u[0] = *(const uint4*)&sB3[n][kc + hlf * 8];
      b3[nt].u[1] = *(const uint4*)&sB3[n][kc + 16 + hlf * 8];
    }
#pragma unroll
    for (int nt = 0; nt < 4; ++nt)
      acc1[nt] = __builtin_amdgcn_wmma_f32_16x16x32_bf16(
          false, fa.v, false, b1[nt].v, (short)0, acc1[nt], false, false);
#pragma unroll
    for (int nt = 0; nt < 4; ++nt)
      acc3[nt] = __builtin_amdgcn_wmma_f32_16x16x32_bf16(
          false, fa.v, false, b3[nt].v, (short)0, acc3[nt], false, false);
  }
}

// -------------------------------------------------------------------------
// GEMM1: h = silu(x @ w1b^T) * (x @ w3b^T)   [T, I], bf16 in/out.
// Double-buffered async stages; inner loop = async DMA || (ds clause + WMMA burst).
// -------------------------------------------------------------------------
__global__ __launch_bounds__(256) void moe_gemm1(const ushort_t* __restrict__ xb,
                                                 const ushort_t* __restrict__ w1b,
                                                 const ushort_t* __restrict__ w3b,
                                                 ushort_t* __restrict__ hb) {
  __shared__ __align__(16) ushort_t sA[2][BM][BK];    // 2 x  8 KB
  __shared__ __align__(16) ushort_t sB1[2][BN][BK];   // 2 x 16 KB
  __shared__ __align__(16) ushort_t sB3[2][BN][BK];   // 2 x 16 KB

  const int tid  = threadIdx.x;
  const int lane = tid & 31;
  const int wave = tid >> 5;
  const int m0   = (wave & 3) * 16;   // wave's M offset (4 M-groups)
  const int nb   = (wave >> 2) * 64;  // wave's N offset (2 N-groups, 4 tiles each)
  const int hlf  = lane >> 4;         // lane half (K-half select per ISA layout)
  const int l16  = lane & 15;
  const int rowBase = blockIdx.x * BM;  // token rows
  const int colBase = blockIdx.y * BN;  // I columns

  const v8f z = {0.f, 0.f, 0.f, 0.f, 0.f, 0.f, 0.f, 0.f};
  v8f acc1[4] = {z, z, z, z};
  v8f acc3[4] = {z, z, z, z};

  stage1(xb, w1b, w3b, rowBase, colBase, 0, tid, sA[0], sB1[0], sB3[0]);
  for (int kk = 0; kk < HID; kk += 2 * BK) {   // 8 iterations, 2 stages each
    publish_stage();                           // stage kk ready in buf0
    stage1(xb, w1b, w3b, rowBase, colBase, kk + BK, tid, sA[1], sB1[1], sB3[1]);
    compute1(sA[0], sB1[0], sB3[0], m0, nb, hlf, l16, acc1, acc3);

    publish_stage();                           // stage kk+BK ready in buf1
    if (kk + 2 * BK < HID)
      stage1(xb, w1b, w3b, rowBase, colBase, kk + 2 * BK, tid, sA[0], sB1[0], sB3[0]);
    compute1(sA[1], sB1[1], sB3[1], m0, nb, hlf, l16, acc1, acc3);
  }

  // Fused SwiGLU epilogue: h = silu(g) * u, store bf16
#pragma unroll
  for (int nt = 0; nt < 4; ++nt) {
#pragma unroll
    for (int r = 0; r < 8; ++r) {
      const int m = m0 + hlf * 8 + r;        // C/D layout: M = r + 8*(lane/16)
      const int n = nb + nt * 16 + l16;
      const float g = acc1[nt][r];
      const float u = acc3[nt][r];
      const float s = g / (1.f + __expf(-g));  // silu
      hb[(size_t)(rowBase + m) * INT_I + colBase + n] = f2bf(s * u);
    }
  }
}

// -------------------------------------------------------------------------
// GEMM2 helpers
// -------------------------------------------------------------------------
__device__ __forceinline__ void stage2(const ushort_t* __restrict__ hb,
                                       const ushort_t* __restrict__ w2b,
                                       int rowBase, int colBase, int kk, int tid,
                                       ushort_t (*sA)[BK], ushort_t (*sB)[BK]) {
#pragma unroll
  for (int i = 0; i < 2; ++i) {
    const int c = tid + i * 256;
    const int r = c >> 3, col = (c & 7) * 8;
    cp16_g2l(hb + (size_t)(rowBase + r) * INT_I + kk + col, &sA[r][col]);
  }
#pragma unroll
  for (int i = 0; i < 4; ++i) {
    const int c = tid + i * 256;
    const int r = c >> 3, col = (c & 7) * 8;
    cp16_g2l(w2b + (size_t)(colBase + r) * INT_I + kk + col, &sB[r][col]);
  }
}

__device__ __forceinline__ void compute2(const ushort_t (*sA)[BK],
                                         const ushort_t (*sB)[BK],
                                         int m0, int nb, int hlf, int l16,
                                         v8f* acc) {
#pragma unroll
  for (int ks = 0; ks < 2; ++ks) {
    const int kc = ks * 32;
    Frag fa;
    fa.u[0] = *(const uint4*)&sA[m0 + l16][kc + hlf * 8];
    fa.u[1] = *(const uint4*)&sA[m0 + l16][kc + 16 + hlf * 8];
    Frag fb[4];
#pragma unroll
    for (int nt = 0; nt < 4; ++nt) {
      const int n = nb + nt * 16 + l16;
      fb[nt].u[0] = *(const uint4*)&sB[n][kc + hlf * 8];
      fb[nt].u[1] = *(const uint4*)&sB[n][kc + 16 + hlf * 8];
    }
#pragma unroll
    for (int nt = 0; nt < 4; ++nt)
      acc[nt] = __builtin_amdgcn_wmma_f32_16x16x32_bf16(
          false, fa.v, false, fb[nt].v, (short)0, acc[nt], false, false);
  }
}

// -------------------------------------------------------------------------
// GEMM2: out += combine[:, e] * (h @ w2b^T)   [T, H], fp32 accumulate.
// Sequential expert launches -> each block exclusively owns its out tile.
// -------------------------------------------------------------------------
__global__ __launch_bounds__(256) void moe_gemm2(const ushort_t* __restrict__ hb,
                                                 const ushort_t* __restrict__ w2b,
                                                 const float* __restrict__ combine,
                                                 int expert,
                                                 float* __restrict__ out) {
  __shared__ __align__(16) ushort_t sA[2][BM][BK];
  __shared__ __align__(16) ushort_t sB[2][BN][BK];

  const int tid  = threadIdx.x;
  const int lane = tid & 31;
  const int wave = tid >> 5;
  const int m0   = (wave & 3) * 16;
  const int nb   = (wave >> 2) * 64;
  const int hlf  = lane >> 4;
  const int l16  = lane & 15;
  const int rowBase = blockIdx.x * BM;  // token rows
  const int colBase = blockIdx.y * BN;  // H columns

  const v8f z = {0.f, 0.f, 0.f, 0.f, 0.f, 0.f, 0.f, 0.f};
  v8f acc[4] = {z, z, z, z};

  stage2(hb, w2b, rowBase, colBase, 0, tid, sA[0], sB[0]);
  for (int kk = 0; kk < INT_I; kk += 2 * BK) {  // 28 iterations, 2 stages each
    publish_stage();
    stage2(hb, w2b, rowBase, colBase, kk + BK, tid, sA[1], sB[1]);
    compute2(sA[0], sB[0], m0, nb, hlf, l16, acc);

    publish_stage();
    if (kk + 2 * BK < INT_I)
      stage2(hb, w2b, rowBase, colBase, kk + 2 * BK, tid, sA[0], sB[0]);
    compute2(sA[1], sB[1], m0, nb, hlf, l16, acc);
  }

#pragma unroll
  for (int nt = 0; nt < 4; ++nt) {
#pragma unroll
    for (int r = 0; r < 8; ++r) {
      const int m = m0 + hlf * 8 + r;
      const int n = nb + nt * 16 + l16;
      const float cw = combine[(size_t)(rowBase + m) * NEXP + expert];
      out[(size_t)(rowBase + m) * HID + colBase + n] += cw * acc[nt][r];
    }
  }
}

// -------------------------------------------------------------------------
// Host-side launcher
// -------------------------------------------------------------------------
extern "C" void kernel_launch(void* const* d_in, const int* in_sizes, int n_in,
                              void* d_out, int out_size, void* d_ws, size_t ws_size,
                              hipStream_t stream) {
  (void)in_sizes; (void)n_in; (void)out_size; (void)ws_size;
  const float* x  = (const float*)d_in[0];  // [T, H]
  const float* gw = (const float*)d_in[1];  // [E, H]
  const float* gb = (const float*)d_in[2];  // [E]
  const float* w1 = (const float*)d_in[3];  // [E, I, H]
  const float* w2 = (const float*)d_in[4];  // [E, H, I]
  const float* w3 = (const float*)d_in[5];  // [E, I, H]
  float* out = (float*)d_out;               // [T, H]

  // Workspace layout (bytes), total ~97.8 MB:
  //   xb:      T*H*2   = 16 MiB   (x in bf16)
  //   hb:      T*I*2   = 56 MiB   (per-expert SwiGLU activations, reused)
  //   combine: T*E*4   = 256 KiB  (dense routing weights)
  //   w1b/w3b/w2b: I*H*2 each = 7 MiB (current expert's weights in bf16)
  char* ws = (char*)d_ws;
  const size_t szXB = (size_t)T_TOK * HID * 2;
  const size_t szHB = (size_t)T_TOK * INT_I * 2;
  const size_t szCB = (size_t)T_TOK * NEXP * 4;
  const size_t szW  = (size_t)INT_I * HID * 2;
  ushort_t* xb      = (ushort_t*)ws;
  ushort_t* hb      = (ushort_t*)(ws + szXB);
  float*    combine = (float*)(ws + szXB + szHB);
  ushort_t* w1b     = (ushort_t*)(ws + szXB + szHB + szCB);
  ushort_t* w3b     = (ushort_t*)(ws + szXB + szHB + szCB + szW);
  ushort_t* w2b     = (ushort_t*)(ws + szXB + szHB + szCB + 2 * szW);

  const int nOut = T_TOK * HID;        // 8.4M
  const int nW8  = (INT_I * HID) / 8;  // 458752 groups of 8

  zero_f32<<<nOut / 256, 256, 0, stream>>>(out, nOut);
  cast_bf16x8<<<(nOut / 8) / 256, 256, 0, stream>>>(x, xb, nOut / 8);
  moe_router<<<T_TOK / 256, 256, 0, stream>>>(x, gw, gb, combine);

  for (int e = 0; e < NEXP; ++e) {
    const size_t wOff = (size_t)e * INT_I * HID;
    cast_bf16x8<<<nW8 / 256, 256, 0, stream>>>(w1 + wOff, w1b, nW8);
    cast_bf16x8<<<nW8 / 256, 256, 0, stream>>>(w3 + wOff, w3b, nW8);
    cast_bf16x8<<<nW8 / 256, 256, 0, stream>>>(w2 + wOff, w2b, nW8);
    moe_gemm1<<<dim3(T_TOK / BM, INT_I / BN), 256, 0, stream>>>(xb, w1b, w3b, hb);
    moe_gemm2<<<dim3(T_TOK / BM, HID / BN), 256, 0, stream>>>(hb, w2b, combine, e, out);
  }
}
